// DeformConv2d_20710332301386
// MI455X (gfx1250) — matrix operand order repaired and checked by reading, not compiled, not run on hardware
//
#include <hip/hip_runtime.h>
#include <hip/hip_bf16.h>

typedef __attribute__((ext_vector_type(16))) _Float16 v16h;
typedef __attribute__((ext_vector_type(8)))  _Float16 v8h;
typedef __attribute__((ext_vector_type(8)))  float    v8f;
typedef __attribute__((ext_vector_type(4)))  unsigned int v4u;
typedef __attribute__((ext_vector_type(4)))  unsigned int u32x4;
typedef __attribute__((ext_vector_type(8)))  int          i32x8;
typedef __attribute__((ext_vector_type(4)))  int          i32x4;

// Problem constants (from reference): B=8, C=256, H=W=64, Cout=256, K=3
// GEMM: M=256 (Cout), N=32768 (B*Ho*Wo), Kdim=2304 (C*9), k = tap*256 + c
#define DIM_M   256
#define DIM_N   32768
#define DIM_K   2304
#define MB      128
#define NB      128
#define KB      32
#define KSTRIDE 40   // KB + 8 f16 pad: 80B rows, 16B aligned, bank-spread

#define USE_TDM __has_builtin(__builtin_amdgcn_tensor_load_to_lds)

// ---------------- Kernel 0: weight fp32 -> f16, layout [Cout][tap*256 + c] ----
__global__ __launch_bounds__(256) void cvt_w(const float* __restrict__ w,
                                             _Float16* __restrict__ A) {
    int i = blockIdx.x * 256 + threadIdx.x;          // over 256*2304
    if (i >= DIM_M * DIM_K) return;
    int o = i / DIM_K;
    int r = i - o * DIM_K;
    int t = r >> 8;          // tap 0..8
    int c = r & 255;         // channel
    A[i] = (_Float16)w[(o * 256 + c) * 9 + t];       // weight[o][c][ky][kx]
}

// ---------------- Kernel 1: bilinear-sample im2col -> col[N][K] f16 ----------
__global__ __launch_bounds__(256) void dcn_im2col(const float* __restrict__ x,
                                                  const float* __restrict__ off,
                                                  _Float16* __restrict__ col) {
    int n = blockIdx.x * 256 + threadIdx.x;          // 0..32767 pixels
    int b  = n >> 12;
    int p  = n & 4095;
    int oh = p >> 6;
    int ow = p & 63;
    const float* xb = x + ((size_t)b << 20);          // b*C*H*W = b*2^20
    const float* ob = off + (size_t)b * 18 * 4096 + p;
    _Float16* crow = col + (size_t)n * DIM_K;

    for (int t = 0; t < 9; ++t) {
        int ky = t / 3, kx = t - ky * 3;
        float py = (float)(oh - 1 + ky) + ob[(size_t)(2 * t) * 4096];
        float px = (float)(ow - 1 + kx) + ob[(size_t)(2 * t + 1) * 4096];
        float yf = floorf(py), xf = floorf(px);
        int   y0 = (int)yf,    x0 = (int)xf;
        float wy = py - yf,    wx = px - xf;
        // fold zero-pad validity into bilinear weights; clamp indices in-bounds
        float vy0 = (y0 >= 0 && y0 < 64) ? 1.f : 0.f;
        float vy1 = (y0 >= -1 && y0 < 63) ? 1.f : 0.f;
        float vx0 = (x0 >= 0 && x0 < 64) ? 1.f : 0.f;
        float vx1 = (x0 >= -1 && x0 < 63) ? 1.f : 0.f;
        int y0c = min(max(y0, 0), 63),     y1c = min(max(y0 + 1, 0), 63);
        int x0c = min(max(x0, 0), 63),     x1c = min(max(x0 + 1, 0), 63);
        float w00 = (1.f - wy) * (1.f - wx) * vy0 * vx0;
        float w01 = (1.f - wy) * wx         * vy0 * vx1;
        float w10 = wy         * (1.f - wx) * vy1 * vx0;
        float w11 = wy         * wx         * vy1 * vx1;
        int i00 = y0c * 64 + x0c, i01 = y0c * 64 + x1c;
        int i10 = y1c * 64 + x0c, i11 = y1c * 64 + x1c;

        _Float16* ct = crow + t * 256;
        for (int c8 = 0; c8 < 32; ++c8) {
            union { v4u u; _Float16 h[8]; } buf;
#pragma unroll
            for (int j = 0; j < 8; ++j) {
                const float* xc = xb + ((size_t)(c8 * 8 + j) << 12); // +c*H*W
                float v = w00 * xc[i00] + w01 * xc[i01]
                        + w10 * xc[i10] + w11 * xc[i11];
                buf.h[j] = (_Float16)v;
            }
            *(v4u*)(ct + c8 * 8) = buf.u;            // 16B packed store
        }
    }
}

#if USE_TDM
// Issue one TDM 2D tile load: 128 rows x 32 f16 from a row-major [rows][2304]
// f16 tensor into LDS, with 4-DWORD pad injected every 16 DWORDs so the LDS
// row stride is 80B (= KSTRIDE f16) for bank-conflict-free fragment reads.
// D# layout per CDNA5 ISA ch.8 (group0 128b + group1 256b; 2D -> groups 2/3 zero).
__device__ __forceinline__ void tdm_load_tile(unsigned lds_off,
                                              const _Float16* gsrc,
                                              unsigned rows_total) {
    unsigned long long ga = (unsigned long long)(size_t)gsrc;
    u32x4 g0 = (u32x4)0u;
    g0.x = 1u;                                         // count=1, user mode
    g0.y = lds_off;                                    // lds_addr [63:32]
    g0.z = (unsigned)(ga & 0xFFFFFFFFull);             // global_addr [95:64]
    g0.w = (unsigned)((ga >> 32) & 0x1FFFFFFull)       // global_addr [120:96]
         | (2u << 30);                                 // type=2 ("image")
    i32x8 g1 = (i32x8)0;
    g1[0] = (1 << 16)                                  // data_size = 2B
          | (1 << 20)                                  // pad_enable
          | (3 << 22)                                  // pad_interval: 16 DWORDs
          | (3 << 25);                                 // pad_amount: 4 DWORDs
    g1[1] = (int)((DIM_K & 0xFFFFu) << 16);            // tensor_dim0 lo16 @48
    g1[2] = (int)((rows_total & 0xFFFFu) << 16);       // tensor_dim1 lo16 @80
    g1[3] = (int)(32u << 16);                          // tile_dim0 = 32 @112
    g1[4] = 128;                                       // tile_dim1=128, tile_dim2=0
    g1[5] = DIM_K;                                     // tensor_dim0_stride lo32
    i32x4 z4 = (i32x4)0;
#if __clang_major__ >= 23
    i32x8 z8 = (i32x8)0;
    __builtin_amdgcn_tensor_load_to_lds(g0, g1, z4, z4, z8, 0);
#else
    __builtin_amdgcn_tensor_load_to_lds(g0, g1, z4, z4, 0);
#endif
}
#endif

// Fragment loads + 8 WMMAs from the given LDS tile buffers.
#define GEMM_COMPUTE(SA, SB)                                                   \
    do {                                                                       \
        union F { v16h v; v8h q[2]; };                                         \
        F a[4], bfr[2];                                                        \
        _Pragma("unroll")                                                      \
        for (int mt = 0; mt < 4; ++mt) {                                       \
            int row = (wm * 64 + mt * 16 + ln) * KSTRIDE;                      \
            a[mt].q[0] = *(const v8h*)&(SA)[row + 8 * h];                      \
            a[mt].q[1] = *(const v8h*)&(SA)[row + 16 + 8 * h];                 \
        }                                                                      \
        _Pragma("unroll")                                                      \
        for (int nt = 0; nt < 2; ++nt) {                                       \
            int row = (wn * 32 + nt * 16 + ln) * KSTRIDE;                      \
            bfr[nt].q[0] = *(const v8h*)&(SB)[row + 8 * h];                    \
            bfr[nt].q[1] = *(const v8h*)&(SB)[row + 16 + 8 * h];               \
        }                                                                      \
        _Pragma("unroll")                                                      \
        for (int mt = 0; mt < 4; ++mt)                                         \
            _Pragma("unroll")                                                  \
            for (int nt = 0; nt < 2; ++nt)                                     \
                acc[mt][nt] = __builtin_amdgcn_wmma_f32_16x16x32_f16(          \
                    false, a[mt].v, false, bfr[nt].v,                          \
                    (short)0, acc[mt][nt], false, false);                      \
    } while (0)

// ---------------- Kernel 2: WMMA GEMM out = A * col^T + bias -----------------
// Block tile 128x128, K-step 32, 8 waves; wave tile 64x32 = 4x2 WMMA tiles.
// TDM path: wave0 issues double-buffered tensor_load_to_lds one K-step ahead;
// s_wait_tensorcnt<=2 keeps the next-buffer fills in flight during WMMA.
__global__ __launch_bounds__(256) void dcn_gemm(const _Float16* __restrict__ A,
                                                const _Float16* __restrict__ Bc,
                                                const float* __restrict__ bias,
                                                float* __restrict__ out) {
    const int tid  = threadIdx.x;
    const int nb   = blockIdx.x & 255;    // 256 N-blocks
    const int mb   = blockIdx.x >> 8;     // 2 M-blocks
    const int wave = tid >> 5, lane = tid & 31;
    const int h    = lane >> 4, ln = lane & 15;
    const int wm   = wave >> 2;           // 0..1  -> 64-row slab
    const int wn   = wave & 3;            // 0..3  -> 32-col slab
    const int KT   = DIM_K / KB;          // 72

    v8f zero = {0.f, 0.f, 0.f, 0.f, 0.f, 0.f, 0.f, 0.f};
    v8f acc[4][2];
#pragma unroll
    for (int mt = 0; mt < 4; ++mt)
#pragma unroll
        for (int nt = 0; nt < 2; ++nt) acc[mt][nt] = zero;

#if USE_TDM
    __shared__ __align__(16) _Float16 sA[2][MB * KSTRIDE];
    __shared__ __align__(16) _Float16 sB[2][NB * KSTRIDE];
    const _Float16* gA = A  + (size_t)(mb * MB) * DIM_K;
    const _Float16* gB = Bc + (size_t)(nb * NB) * DIM_K;
    // LDS byte offsets: generic shared-aperture addresses carry the LDS offset
    // in their low 32 bits (ISA 10.2 aperture mapping).
    unsigned ldsA[2] = { (unsigned)(size_t)(void*)&sA[0][0],
                         (unsigned)(size_t)(void*)&sA[1][0] };
    unsigned ldsB[2] = { (unsigned)(size_t)(void*)&sB[0][0],
                         (unsigned)(size_t)(void*)&sB[1][0] };

    if (wave == 0) {                      // prologue fill of buffer 0
        tdm_load_tile(ldsA[0], gA, DIM_M);
        tdm_load_tile(ldsB[0], gB, DIM_N);
    }
    for (int kt = 0; kt < KT; ++kt) {
        const int cur = kt & 1;
        if (wave == 0) {
            if (kt + 1 < KT) {            // prefetch next K-slab into other buf
                tdm_load_tile(ldsA[cur ^ 1], gA + (kt + 1) * KB, DIM_M);
                tdm_load_tile(ldsB[cur ^ 1], gB + (kt + 1) * KB, DIM_N);
                __builtin_amdgcn_s_wait_tensorcnt(2);  // cur fills complete
            } else {
                __builtin_amdgcn_s_wait_tensorcnt(0);
            }
        }
        __syncthreads();                  // broadcast cur-buffer readiness
        GEMM_COMPUTE(sA[cur], sB[cur]);
        __syncthreads();                  // all reads of cur done -> refillable
    }
#else
    // Fallback: register-staged global->LDS (single buffer, reg prefetch)
    __shared__ __align__(16) _Float16 sA[MB * KSTRIDE];
    __shared__ __align__(16) _Float16 sB[NB * KSTRIDE];
    const int r0 = tid >> 2;              // row 0..63
    const int s0 = tid & 3;               // 16B segment within 64B row
    const _Float16* gA0 = A  + (size_t)(mb * MB + r0) * DIM_K + s0 * 8;
    const _Float16* gA1 = gA0 + (size_t)64 * DIM_K;
    const _Float16* gB0 = Bc + (size_t)(nb * NB + r0) * DIM_K + s0 * 8;
    const _Float16* gB1 = gB0 + (size_t)64 * DIM_K;
    _Float16* lA0 = &sA[r0 * KSTRIDE + s0 * 8];
    _Float16* lA1 = lA0 + 64 * KSTRIDE;
    _Float16* lB0 = &sB[r0 * KSTRIDE + s0 * 8];
    _Float16* lB1 = lB0 + 64 * KSTRIDE;
    v4u pa0 = *(const v4u*)gA0, pa1 = *(const v4u*)gA1;
    v4u pb0 = *(const v4u*)gB0, pb1 = *(const v4u*)gB1;
    for (int kt = 0; kt < KT; ++kt) {
        *(v4u*)lA0 = pa0; *(v4u*)lA1 = pa1;
        *(v4u*)lB0 = pb0; *(v4u*)lB1 = pb1;
        __syncthreads();
        if (kt + 1 < KT) {
            int k0 = (kt + 1) * KB;
            pa0 = *(const v4u*)(gA0 + k0); pa1 = *(const v4u*)(gA1 + k0);
            pb0 = *(const v4u*)(gB0 + k0); pb1 = *(const v4u*)(gB1 + k0);
        }
        GEMM_COMPUTE(sA, sB);
        __syncthreads();
    }
#endif

    // Epilogue: D lane map m = r + 8h (VGPR r), n = lane&15. out[b][m][p]
    const int mBase = mb * MB + wm * 64;
#pragma unroll
    for (int nt = 0; nt < 2; ++nt) {
        int ncol = nb * NB + wn * 32 + nt * 16 + ln;
        int bi   = ncol >> 12;
        int p    = ncol & 4095;
        float* op = out + ((size_t)bi * 256 + mBase) * 4096 + p;
#pragma unroll
        for (int mt = 0; mt < 4; ++mt)
#pragma unroll
            for (int r = 0; r < 8; ++r) {
                int mrel = mt * 16 + 8 * h + r;
                op[(size_t)mrel * 4096] = acc[mt][nt][r] + bias[mBase + mrel];
            }
    }
}

extern "C" void kernel_launch(void* const* d_in, const int* in_sizes, int n_in,
                              void* d_out, int out_size, void* d_ws, size_t ws_size,
                              hipStream_t stream) {
    const float* x    = (const float*)d_in[0];   // [8,256,64,64]
    const float* off  = (const float*)d_in[1];   // [8,18,64,64]
    const float* w    = (const float*)d_in[2];   // [256,256,3,3]
    const float* bias = (const float*)d_in[3];   // [256]
    float* out = (float*)d_out;                  // [8,256,64,64] f32

    // workspace: col f16 [32768][2304] = 150,994,944 B (fits MI455X 192MB L2),
    // then A f16 [256][2304] = 1,179,648 B
    _Float16* col = (_Float16*)d_ws;
    _Float16* A   = (_Float16*)((char*)d_ws + (size_t)DIM_N * DIM_K * 2);

    cvt_w<<<(DIM_M * DIM_K + 255) / 256, 256, 0, stream>>>(w, A);
    dcn_im2col<<<DIM_N / 256, 256, 0, stream>>>(x, off, col);
    dcn_gemm<<<(DIM_M / MB) * (DIM_N / NB), 256, 0, stream>>>(A, col, bias, out);
}
